// ManualCrossAttention_7851200217618
// MI455X (gfx1250) — matrix operand order repaired and checked by reading, not compiled, not run on hardware
//
#include <hip/hip_runtime.h>

// Problem constants (reference: B=32, P=512, S=1024, E=512, H=8, D=64)
#define B_ 32
#define P_ 512
#define S_ 1024
#define E_ 512
#define H_ 8
#define D_ 64

typedef __attribute__((ext_vector_type(16))) __bf16 v16bf;
typedef __attribute__((ext_vector_type(8)))  __bf16 v8bf;
typedef __attribute__((ext_vector_type(8)))  float  v8f;

__device__ __forceinline__ __bf16 f2bf(float x) { return (__bf16)x; }

// D = A(16x32 bf16) * B(32x16 bf16) + C(16x16 f32), wave32
__device__ __forceinline__ v8f wmma_bf16(v16bf a, v16bf b, v8f c) {
    return __builtin_amdgcn_wmma_f32_16x16x32_bf16(
        /*neg_a=*/false, a, /*neg_b=*/false, b,
        /*c_mod=*/(short)0, c, /*reuse_a=*/false, /*reuse_b=*/false);
}

// A-fragment K index for VGPR i (contiguous order, matches B loads):
// lanes 0-15: VGPR0..7 -> K={0..15}; lanes 16-31: K={16..31}
__device__ __forceinline__ int a_k_of(int i, int khalf) {
    return 2 * i + 2 * khalf;
}

// ---------------------------------------------------------------------------
// 1) fp32 -> bf16 conversion (weights)
// ---------------------------------------------------------------------------
__global__ void cvt_f32_bf16(const float* __restrict__ in, __bf16* __restrict__ out, int n) {
    int i = blockIdx.x * blockDim.x + threadIdx.x;
    if (i < n) out[i] = f2bf(in[i]);
}

// ---------------------------------------------------------------------------
// 2) Projection GEMM:  out[b, row, col] = sum_k X[b,row,k] * W[col,k]
//    X fp32 [B, L, E] (converted to bf16 on the fly), W bf16 [E, E] row-major.
//    Each wave computes a 16x64 output tile (4 WMMA accumulators); the tile is
//    64-aligned in columns so it spans exactly one head: h = tn, d = j*16+lane.
//    TRANSPOSED==0 : write [B, H, L, D]  (Q, K) -- strided b16 stores
//    TRANSPOSED==1 : write [B, H, D, L]  (V->Vt) -- 8 rows contiguous, one
//                    packed b128 store per sub-tile.
// ---------------------------------------------------------------------------
template <int TRANSPOSED>
__global__ void __launch_bounds__(256)
proj_bf16_kernel(const float* __restrict__ X, const __bf16* __restrict__ W,
                 __bf16* __restrict__ out, int L) {
    const int wave = threadIdx.x >> 5;
    const int lane = threadIdx.x & 31;
    const int tile = blockIdx.x * 8 + wave;

    const int tiles_n = E_ / 64;           // 8 column-tiles of width 64
    const int tiles_m = L / 16;
    const int b   = tile / (tiles_m * tiles_n);
    const int rem = tile - b * tiles_m * tiles_n;
    const int tm  = rem / tiles_n;
    const int tn  = rem - tm * tiles_n;    // == head index h

    const int mlane = lane & 15;
    const int khalf = (lane & 16) ? 8 : 0;
    const int kh16  = (lane & 16) ? 16 : 0;

    const float*  arow  = X + ((size_t)b * L + (tm * 16 + mlane)) * E_;
    const __bf16* brow0 = W + (size_t)(tn * 64 + mlane) * E_ + kh16;

    v8f acc[4] = {{}, {}, {}, {}};
    for (int kk = 0; kk < E_; kk += 32) {
        __builtin_prefetch(arow + kk + 64, 0, 1);   // global_prefetch_b8
        v16bf a;
#pragma unroll
        for (int i = 0; i < 8; ++i) {
            int kb = kk + a_k_of(i, khalf);
            a[2 * i]     = f2bf(arow[kb]);
            a[2 * i + 1] = f2bf(arow[kb + 1]);
        }
        v16bf bfr[4];
#pragma unroll
        for (int j = 0; j < 4; ++j)
            bfr[j] = *(const v16bf*)(brow0 + (size_t)(j * 16) * E_ + kk);
#pragma unroll
        for (int j = 0; j < 4; ++j)
            acc[j] = wmma_bf16(a, bfr[j], acc[j]);
    }

    const int h     = tn;
    const int mbase = (lane & 16) ? 8 : 0;
    const int row0  = tm * 16 + mbase;

    if (TRANSPOSED) {
        // out[((b*H + h)*D + d)*L + row] ; 8 rows contiguous -> packed b128
        __bf16* obase = out + (((size_t)b * H_ + h) * D_) * L + row0;
#pragma unroll
        for (int j = 0; j < 4; ++j) {
            const int d = j * 16 + mlane;
            v8bf pk;
#pragma unroll
            for (int r = 0; r < 8; ++r) pk[r] = f2bf(acc[j][r]);
            *(v8bf*)(obase + (size_t)d * L) = pk;   // 16B aligned
        }
    } else {
        // out[((b*H + h)*L + row)*D + d] ; stride D between rows
        __bf16* obase = out + (((size_t)b * H_ + h) * L + row0) * D_ + mlane;
#pragma unroll
        for (int j = 0; j < 4; ++j) {
#pragma unroll
            for (int r = 0; r < 8; ++r)
                obase[(size_t)r * D_ + j * 16] = f2bf(acc[j][r]);
        }
    }
}

// ---------------------------------------------------------------------------
// 3) Fused attention: one workgroup = (b, h, 16-row P tile), 4 waves.
//    Phase 1: scores = Q K^T * scale, masked, into LDS (16 x 1024 fp32 = 64KB)
//    Phase 2: row softmax, normalized in place (shfl reductions, no extra LDS)
//    Phase 3: ctx = P @ V via WMMA; each wave owns a 16-wide D tile
// ---------------------------------------------------------------------------
__global__ void __launch_bounds__(128)
attn_kernel(const __bf16* __restrict__ Qp,  // [B,H,P,D] bf16
            const __bf16* __restrict__ Kp,  // [B,H,S,D] bf16
            const __bf16* __restrict__ Vt,  // [B,H,D,S] bf16
            const unsigned char* __restrict__ mask,  // [B,P,S] bool
            __bf16* __restrict__ ctx)       // [B,P,E] bf16
{
    __shared__ float sc[16 * S_];  // exactly 64 KB

    const int ptile = blockIdx.x & 31;   // P_/16 == 32
    const int bh    = blockIdx.x >> 5;
    const int h     = bh & (H_ - 1);
    const int b     = bh / H_;

    const int wave  = threadIdx.x >> 5;
    const int lane  = threadIdx.x & 31;
    const int mlane = lane & 15;
    const int khalf = (lane & 16) ? 8 : 0;
    const int kh16  = (lane & 16) ? 16 : 0;
    const int mbase = (lane & 16) ? 8 : 0;

    const float scale = 0.125f;  // 1/sqrt(D)

    // ---- Phase 1: scores -> LDS, scale+mask fused into writeback ----
    {
        const __bf16* qrow =
            Qp + (((size_t)b * H_ + h) * P_ + ptile * 16 + mlane) * D_;
        // Q A-fragments (K = 0..31 and 32..63) reused across all 64 S tiles
        v16bf aq0, aq1;
#pragma unroll
        for (int i = 0; i < 8; ++i) {
            int k0 = a_k_of(i, khalf);
            aq0[2 * i]     = qrow[k0];
            aq0[2 * i + 1] = qrow[k0 + 1];
            aq1[2 * i]     = qrow[32 + k0];
            aq1[2 * i + 1] = qrow[32 + k0 + 1];
        }
        for (int st = wave; st < S_ / 16; st += 4) {
            const __bf16* krow =
                Kp + (((size_t)b * H_ + h) * S_ + st * 16 + mlane) * D_;
            v8f acc = {};
            acc = wmma_bf16(aq0, *(const v16bf*)(krow + kh16), acc);
            acc = wmma_bf16(aq1, *(const v16bf*)(krow + 32 + kh16), acc);

            const int scol = st * 16 + mlane;
            const unsigned char* mcol =
                mask + ((size_t)b * P_ + ptile * 16 + mbase) * S_ + scol;
            float* scp = sc + mbase * S_ + scol;
#pragma unroll
            for (int r = 0; r < 8; ++r) {
                scp[r * S_] =
                    mcol[(size_t)r * S_] ? -3.0e38f : acc[r] * scale;
            }
        }
    }
    __syncthreads();

    // ---- Phase 2: softmax; 8 lanes per row, shuffle reductions ----
    {
        const int row = threadIdx.x >> 3;               // 0..15
        float* r = sc + row * S_ + (threadIdx.x & 7) * (S_ / 8);
        float lmax = -3.4e38f;
        for (int c = 0; c < S_ / 8; ++c) lmax = fmaxf(lmax, r[c]);
#pragma unroll
        for (int o = 1; o < 8; o <<= 1) lmax = fmaxf(lmax, __shfl_xor(lmax, o, 8));
        float lsum = 0.f;
        for (int c = 0; c < S_ / 8; ++c) {
            float e = __expf(r[c] - lmax);
            r[c] = e;
            lsum += e;
        }
#pragma unroll
        for (int o = 1; o < 8; o <<= 1) lsum += __shfl_xor(lsum, o, 8);
        float inv = 1.0f / fmaxf(lsum, 1e-30f);
        for (int c = 0; c < S_ / 8; ++c) r[c] *= inv;   // normalize in place
    }
    __syncthreads();

    // ---- Phase 3: ctx = P @ V ; wave owns d-tile [wave*16, wave*16+16) ----
    {
        const float*  prow = sc + mlane * S_;
        const __bf16* vrow =
            Vt + (((size_t)b * H_ + h) * D_ + wave * 16 + mlane) * S_;
        v8f acc = {};
        for (int kk = 0; kk < S_; kk += 32) {
            v16bf a;
#pragma unroll
            for (int i = 0; i < 8; ++i) {
                int kb = kk + a_k_of(i, khalf);
                a[2 * i]     = f2bf(prow[kb]);
                a[2 * i + 1] = f2bf(prow[kb + 1]);
            }
            v16bf bf = *(const v16bf*)(vrow + kk + kh16);
            acc = wmma_bf16(a, bf, acc);
        }
        const int d = wave * 16 + mlane;
        __bf16* obase =
            ctx + ((size_t)b * P_ + ptile * 16 + mbase) * E_ + h * D_ + d;
#pragma unroll
        for (int r = 0; r < 8; ++r)
            obase[(size_t)r * E_] = f2bf(acc[r]);
    }
}

// ---------------------------------------------------------------------------
// 4) Output GEMM: out[row, col] = sum_k ctx[row,k] * Wo[col,k] + bo[col]
//    ctx bf16 [B*P, E], out fp32. 16x64 tile per wave (4 accumulators).
// ---------------------------------------------------------------------------
__global__ void __launch_bounds__(256)
out_gemm_kernel(const __bf16* __restrict__ ctx, const __bf16* __restrict__ Wo,
                const float* __restrict__ bo, float* __restrict__ out) {
    const int wave = threadIdx.x >> 5;
    const int lane = threadIdx.x & 31;
    const int tile = blockIdx.x * 8 + wave;

    const int tiles_n = E_ / 64;       // 8
    const int tm = tile / tiles_n;     // 0..(B*P/16 - 1)
    const int tn = tile - tm * tiles_n;

    const int mlane = lane & 15;
    const int khalf = (lane & 16) ? 8 : 0;
    const int kh16  = (lane & 16) ? 16 : 0;

    const __bf16* arow  = ctx + (size_t)(tm * 16 + mlane) * E_;
    const __bf16* brow0 = Wo  + (size_t)(tn * 64 + mlane) * E_ + kh16;

    v8f acc[4] = {{}, {}, {}, {}};
    for (int kk = 0; kk < E_; kk += 32) {
        __builtin_prefetch(arow + kk + 32, 0, 1);   // global_prefetch_b8
        v16bf a;
#pragma unroll
        for (int i = 0; i < 8; ++i) {
            int kb = kk + a_k_of(i, khalf);
            a[2 * i]     = arow[kb];
            a[2 * i + 1] = arow[kb + 1];
        }
        v16bf bfr[4];
#pragma unroll
        for (int j = 0; j < 4; ++j)
            bfr[j] = *(const v16bf*)(brow0 + (size_t)(j * 16) * E_ + kk);
#pragma unroll
        for (int j = 0; j < 4; ++j)
            acc[j] = wmma_bf16(a, bfr[j], acc[j]);
    }

    const int mbase = (lane & 16) ? 8 : 0;
    float* obase = out + (size_t)(tm * 16 + mbase) * E_ + tn * 64 + mlane;
#pragma unroll
    for (int j = 0; j < 4; ++j) {
        const float bias = bo[tn * 64 + j * 16 + mlane];
#pragma unroll
        for (int r = 0; r < 8; ++r)
            obase[(size_t)r * E_ + j * 16] = acc[j][r] + bias;
    }
}

// ---------------------------------------------------------------------------
extern "C" void kernel_launch(void* const* d_in, const int* in_sizes, int n_in,
                              void* d_out, int out_size, void* d_ws, size_t ws_size,
                              hipStream_t stream) {
    const float* query = (const float*)d_in[0];   // [B,P,E]
    const float* keyi  = (const float*)d_in[1];   // [B,S,E]
    const float* valu  = (const float*)d_in[2];   // [B,S,E]
    const unsigned char* mask = (const unsigned char*)d_in[3];  // [B,P,S] bool
    const float* Wq = (const float*)d_in[4];
    const float* Wk = (const float*)d_in[5];
    const float* Wv = (const float*)d_in[6];
    const float* Wo = (const float*)d_in[7];
    const float* bo = (const float*)d_in[8];
    float* out = (float*)d_out;                   // [B,P,E] fp32

    // Workspace layout (~103 MB)
    char*  ws  = (char*)d_ws;
    size_t off = 0;
    __bf16* Wq_bf = (__bf16*)(ws + off); off += (size_t)E_ * E_ * 2;
    __bf16* Wk_bf = (__bf16*)(ws + off); off += (size_t)E_ * E_ * 2;
    __bf16* Wv_bf = (__bf16*)(ws + off); off += (size_t)E_ * E_ * 2;
    __bf16* Wo_bf = (__bf16*)(ws + off); off += (size_t)E_ * E_ * 2;
    __bf16* Qp   = (__bf16*)(ws + off); off += (size_t)B_ * H_ * P_ * D_ * 2;
    __bf16* Kp   = (__bf16*)(ws + off); off += (size_t)B_ * H_ * S_ * D_ * 2;
    __bf16* Vt   = (__bf16*)(ws + off); off += (size_t)B_ * H_ * S_ * D_ * 2;
    __bf16* ctxb = (__bf16*)(ws + off); off += (size_t)B_ * P_ * E_ * 2;

    // 1) weights fp32 -> bf16 (once; reused by every row tile)
    const int nw = E_ * E_;
    cvt_f32_bf16<<<(nw + 255) / 256, 256, 0, stream>>>(Wq, Wq_bf, nw);
    cvt_f32_bf16<<<(nw + 255) / 256, 256, 0, stream>>>(Wk, Wk_bf, nw);
    cvt_f32_bf16<<<(nw + 255) / 256, 256, 0, stream>>>(Wv, Wv_bf, nw);
    cvt_f32_bf16<<<(nw + 255) / 256, 256, 0, stream>>>(Wo, Wo_bf, nw);

    // 2) projections (8 waves / block, one 16x64 tile / wave; exact grids)
    proj_bf16_kernel<0><<<B_ * (P_ / 16) * (E_ / 64) / 8, 256, 0, stream>>>(
        query, Wq_bf, Qp, P_);
    proj_bf16_kernel<0><<<B_ * (S_ / 16) * (E_ / 64) / 8, 256, 0, stream>>>(
        keyi, Wk_bf, Kp, S_);
    proj_bf16_kernel<1><<<B_ * (S_ / 16) * (E_ / 64) / 8, 256, 0, stream>>>(
        valu, Wv_bf, Vt, S_);

    // 3) fused scores + softmax + ctx
    attn_kernel<<<B_ * H_ * (P_ / 16), 128, 0, stream>>>(Qp, Kp, Vt, mask, ctxb);

    // 4) output projection + bias
    out_gemm_kernel<<<(B_ * P_ / 16) * (E_ / 64) / 8, 256, 0, stream>>>(
        ctxb, Wo_bf, bo, out);
}
